// GRUAggregator_78941498901090
// MI455X (gfx1250) — matrix operand order repaired and checked by reading, not compiled, not run on hardware
//
#include <hip/hip_runtime.h>
#include <stdint.h>

#define D 256
#define HS 20      // padded LDS row stride for H (floats): avoids bank conflicts
#define GROUP 16   // nodes per recurrence workgroup

typedef __attribute__((ext_vector_type(16))) _Float16 v16h;
typedef __attribute__((ext_vector_type(8)))  float    v8f;

union H2 { uint32_t u; _Float16 h[2]; };

// ---------------- phase 1: counting sort by node, then sort-by-t ----------------

__global__ void k_zero(int* p, int n) {
    int i = blockIdx.x * blockDim.x + threadIdx.x;
    if (i < n) p[i] = 0;
}

__global__ void k_count(const int* __restrict__ idx, int* __restrict__ counts, int E) {
    int e = blockIdx.x * blockDim.x + threadIdx.x;
    if (e < E) atomicAdd(&counts[idx[e]], 1);
}

// single block, 1024 threads: exclusive scan counts[0..N) -> segs[0..N]
__global__ void k_scan(const int* __restrict__ counts, int* __restrict__ segs, int N) {
    __shared__ int part[1024];
    const int t = threadIdx.x;
    const int ipt = (N + 1023) >> 10;      // items per thread (<=16)
    const int base = t * ipt;
    int local[16];
    int s = 0;
    for (int j = 0; j < ipt; ++j) {
        int i = base + j;
        local[j] = s;
        s += (i < N) ? counts[i] : 0;
    }
    part[t] = s;
    __syncthreads();
    for (int off = 1; off < 1024; off <<= 1) {
        int v = (t >= off) ? part[t - off] : 0;
        __syncthreads();
        part[t] += v;
        __syncthreads();
    }
    int excl = (t == 0) ? 0 : part[t - 1];
    for (int j = 0; j < ipt; ++j) {
        int i = base + j;
        if (i < N) segs[i] = excl + local[j];
    }
    if (t == 1023) segs[N] = part[1023];
}

__global__ void k_scatter(const int* __restrict__ idx, const int* __restrict__ segs,
                          int* __restrict__ cursor, int* __restrict__ order, int E) {
    int e = blockIdx.x * blockDim.x + threadIdx.x;
    if (e < E) {
        int n = idx[e];
        int p = segs[n] + atomicAdd(&cursor[n], 1);
        order[p] = e;
    }
}

// one thread per node: in-place insertion sort of its segment by key t
__global__ void k_sort(const float* __restrict__ tkey, const int* __restrict__ segs,
                       int* __restrict__ order, int N) {
    int n = blockIdx.x * blockDim.x + threadIdx.x;
    if (n >= N) return;
    int s = segs[n], e = segs[n + 1];
    for (int i = s + 1; i < e; ++i) {
        int v = order[i];
        float key = tkey[v];
        int j = i - 1;
        while (j >= s) {
            int w = order[j];
            if (tkey[w] <= key) break;
            order[j + 1] = w;
            --j;
        }
        order[j + 1] = v;
    }
}

// ---------------- phase 2: batched WMMA recurrence ----------------
// One workgroup = 16 nodes. Per step: X = (m + H) as f16 (B layout in LDS),
// Y = W @ X + b via v_wmma_f32_16x16x32_f16, write active columns back to H.

__global__ __launch_bounds__(256)
void k_recur(const float* __restrict__ msg, const int* __restrict__ segs,
             const int* __restrict__ order, const float* __restrict__ W,
             const float* __restrict__ bias, float* __restrict__ out) {
    __shared__ float    Hs[D * HS];     // H[row][col], padded stride
    __shared__ uint32_t Xs[8 * 256];    // X in B-operand layout: [kt][v*32 + lane]
    __shared__ int sseg[GROUP];
    __shared__ int slen[GROUP];

    const int tid  = threadIdx.x;
    const int lane = tid & 31;
    const int wv   = tid >> 5;          // wave 0..7; also the kt this thread fills in X
    const int nodeBase = blockIdx.x * GROUP;

    if (tid < GROUP) {
        int s = segs[nodeBase + tid];
        int e = segs[nodeBase + tid + 1];
        sseg[tid] = s;
        slen[tid] = e - s;
    }
    for (int j = tid; j < D * HS; j += 256) Hs[j] = 0.0f;

    // ---- load W tiles (f32 -> f16) into registers, A-operand layout ----
    // A 16x32 f16 layout: lanes 0-15 = M rows, half = lane>>4 selects K-subblocks.
    const int half = lane >> 4;
    const int mrow = lane & 15;
    v16h wA[2][8];
#pragma unroll
    for (int mi = 0; mi < 2; ++mi) {
        const int mt = 2 * wv + mi;
        const float* wr = W + (size_t)(mt * 16 + mrow) * D;
#pragma unroll
        for (int kt = 0; kt < 8; ++kt) {
            const int k0 = kt * 32 + half * 16;
            float4 w0 = *(const float4*)(wr + k0);
            float4 w1 = *(const float4*)(wr + k0 + 4);
            float4 w2 = *(const float4*)(wr + k0 + 8);
            float4 w3 = *(const float4*)(wr + k0 + 12);
            v16h a;
            a[0]  = (_Float16)w0.x; a[1]  = (_Float16)w0.y;
            a[2]  = (_Float16)w0.z; a[3]  = (_Float16)w0.w;
            a[4]  = (_Float16)w1.x; a[5]  = (_Float16)w1.y;
            a[6]  = (_Float16)w1.z; a[7]  = (_Float16)w1.w;
            a[8]  = (_Float16)w2.x; a[9]  = (_Float16)w2.y;
            a[10] = (_Float16)w2.z; a[11] = (_Float16)w2.w;
            a[12] = (_Float16)w3.x; a[13] = (_Float16)w3.y;
            a[14] = (_Float16)w3.z; a[15] = (_Float16)w3.w;
            wA[mi][kt] = a;
        }
    }
    // bias in accumulator (C/D) layout: element (m,n) -> vgpr r = m&7, m = half*8+r
    float binit[2][8];
#pragma unroll
    for (int mi = 0; mi < 2; ++mi) {
        const int mt = 2 * wv + mi;
#pragma unroll
        for (int r = 0; r < 8; ++r) binit[mi][r] = bias[mt * 16 + half * 8 + r];
    }
    __syncthreads();

    // X-building mapping: thread handles column n, rows kb..kb+15
    const int n  = tid & 15;
    const int kb = (tid >> 4) * 16;
    const int myS   = sseg[n];
    const int myLen = slen[n];

    int maxLen = 0;
    for (int i = 0; i < GROUP; ++i) maxLen = max(maxLen, slen[i]);

    for (int k = 0; k < maxLen; ++k) {
        __syncthreads();  // prev-iter H writes visible; prev X fully consumed
        const bool act = (k < myLen);
        const float* mp = act ? (msg + (size_t)order[myS + k] * D + kb) : msg;
        uint32_t pk[8];
#pragma unroll
        for (int q = 0; q < 4; ++q) {
            float4 mv = act ? *(const float4*)(mp + 4 * q) : make_float4(0.f, 0.f, 0.f, 0.f);
            float x0 = mv.x + Hs[(kb + 4 * q + 0) * HS + n];
            float x1 = mv.y + Hs[(kb + 4 * q + 1) * HS + n];
            float x2 = mv.z + Hs[(kb + 4 * q + 2) * HS + n];
            float x3 = mv.w + Hs[(kb + 4 * q + 3) * HS + n];
            H2 p0, p1;
            p0.h[0] = (_Float16)x0; p0.h[1] = (_Float16)x1;
            p1.h[0] = (_Float16)x2; p1.h[1] = (_Float16)x3;
            pk[2 * q]     = p0.u;
            pk[2 * q + 1] = p1.u;
        }
        const int xbase = wv * 256 + lane;  // kt == wv for this thread's rows
#pragma unroll
        for (int v = 0; v < 8; ++v) Xs[xbase + v * 32] = pk[v];
        __syncthreads();

        v8f acc0, acc1;
#pragma unroll
        for (int r = 0; r < 8; ++r) { acc0[r] = binit[0][r]; acc1[r] = binit[1][r]; }
#pragma unroll
        for (int kt = 0; kt < 8; ++kt) {
            v16h bm;
            H2 tmp;
#pragma unroll
            for (int v = 0; v < 8; ++v) {
                tmp.u = Xs[kt * 256 + v * 32 + lane];
                bm[2 * v]     = tmp.h[0];
                bm[2 * v + 1] = tmp.h[1];
            }
            acc0 = __builtin_amdgcn_wmma_f32_16x16x32_f16(
                false, wA[0][kt], false, bm, (short)0, acc0, false, false);
            acc1 = __builtin_amdgcn_wmma_f32_16x16x32_f16(
                false, wA[1][kt], false, bm, (short)0, acc1, false, false);
        }
        // write back active columns (this lane's column == n)
        if (act) {
#pragma unroll
            for (int r = 0; r < 8; ++r) {
                Hs[((2 * wv + 0) * 16 + half * 8 + r) * HS + n] = acc0[r];
                Hs[((2 * wv + 1) * 16 + half * 8 + r) * HS + n] = acc1[r];
            }
        }
    }
    __syncthreads();

    float* orow = out + (size_t)(nodeBase + n) * D + kb;
#pragma unroll
    for (int q = 0; q < 4; ++q) {
        float4 o;
        o.x = Hs[(kb + 4 * q + 0) * HS + n];
        o.y = Hs[(kb + 4 * q + 1) * HS + n];
        o.z = Hs[(kb + 4 * q + 2) * HS + n];
        o.w = Hs[(kb + 4 * q + 3) * HS + n];
        *(float4*)(orow + 4 * q) = o;
    }
}

// ---------------- launcher ----------------

extern "C" void kernel_launch(void* const* d_in, const int* in_sizes, int n_in,
                              void* d_out, int out_size, void* d_ws, size_t ws_size,
                              hipStream_t stream) {
    const float* msg  = (const float*)d_in[0];
    const int*   idx  = (const int*)d_in[1];
    const float* tkey = (const float*)d_in[2];
    const float* W    = (const float*)d_in[4];
    const float* bias = (const float*)d_in[5];

    const int E = in_sizes[1];
    const int N = out_size / D;   // dim_size (output is N x D)

    int* ws     = (int*)d_ws;
    int* counts = ws;              // N
    int* cursor = ws + N;          // N
    int* segs   = ws + 2 * N;      // N + 1
    int* order  = ws + 3 * N + 8;  // E

    const int zn = 2 * N;  // zero counts + cursor
    k_zero   <<<(zn + 255) / 256, 256, 0, stream>>>(ws, zn);
    k_count  <<<(E + 255) / 256, 256, 0, stream>>>(idx, counts, E);
    k_scan   <<<1, 1024, 0, stream>>>(counts, segs, N);
    k_scatter<<<(E + 255) / 256, 256, 0, stream>>>(idx, segs, cursor, order, E);
    k_sort   <<<(N + 255) / 256, 256, 0, stream>>>(tkey, segs, order, N);
    k_recur  <<<N / GROUP, 256, 0, stream>>>(msg, segs, order, W, bias, (float*)d_out);
}
